// UBlock_2963527434275
// MI455X (gfx1250) — compile-verified, matrix operand order checked by reading, not run on hardware
//
#include <hip/hip_runtime.h>

// ---------------------------------------------------------------------------
// CDNA5 (gfx1250, wave32) sparse-conv UNet.
//   conv  = rulebook gather -> bf16 WMMA (f32 accumulate) -> f32 atomic scatter
//   bn    = column sum/sumsq reduction + fused apply (+residual +relu +bf16 out)
// ---------------------------------------------------------------------------

typedef __attribute__((ext_vector_type(16))) __bf16 v16bf;
typedef __attribute__((ext_vector_type(8)))  __bf16 v8bf;
typedef __attribute__((ext_vector_type(8)))  float  v8f;

#define EPSV 1e-5f

// ---------------- utility kernels ----------------

__global__ void zero_f32_kernel(float* __restrict__ p, size_t n) {
  size_t i = (size_t)blockIdx.x * blockDim.x + threadIdx.x;
  size_t stride = (size_t)gridDim.x * blockDim.x;
  for (; i < n; i += stride) p[i] = 0.f;
}

__global__ void iota_kernel(int* __restrict__ p, int n) {
  int i = blockIdx.x * blockDim.x + threadIdx.x;
  if (i < n) p[i] = i;
}

// f32 (n x C) -> bf16 rows [0..n) at (row*ystride + yoff), row n zero-padded.
__global__ void cvt_bf_kernel(const float* __restrict__ x, __bf16* __restrict__ y,
                              int n, int C, int ystride, int yoff) {
  size_t i = (size_t)blockIdx.x * blockDim.x + threadIdx.x;
  size_t total = (size_t)(n + 1) * C;
  if (i >= total) return;
  int row = (int)(i / C), c = (int)(i % C);
  y[(size_t)row * ystride + yoff + c] = (row == n) ? (__bf16)0.f : (__bf16)x[i];
}

// one block per column: stats[c] = sum, stats[C+c] = sum of squares
__global__ void bn_stats_kernel(const float* __restrict__ x, int n, int C,
                                float* __restrict__ stats) {
  int c = blockIdx.x;
  float s = 0.f, s2 = 0.f;
  for (int i = threadIdx.x; i < n; i += blockDim.x) {
    float v = x[(size_t)i * C + c];
    s += v; s2 += v * v;
  }
  __shared__ float sh0[256], sh1[256];
  sh0[threadIdx.x] = s; sh1[threadIdx.x] = s2;
  __syncthreads();
  for (int o = 128; o > 0; o >>= 1) {
    if ((int)threadIdx.x < o) {
      sh0[threadIdx.x] += sh0[threadIdx.x + o];
      sh1[threadIdx.x] += sh1[threadIdx.x + o];
    }
    __syncthreads();
  }
  if (threadIdx.x == 0) { stats[c] = sh0[0]; stats[C + c] = sh1[0]; }
}

// y = relu?( bn(x)*g + b (+resid) ); optional f32 out and/or padded bf16 out
__global__ void bn_apply_kernel(const float* __restrict__ x, const float* __restrict__ stats,
                                const float* __restrict__ g, const float* __restrict__ b,
                                const float* __restrict__ resid, float* __restrict__ yf,
                                __bf16* __restrict__ ybf, int n, int C, int bstride, int boff,
                                int relu, float invn) {
  size_t i = (size_t)blockIdx.x * blockDim.x + threadIdx.x;
  size_t total = (size_t)(n + 1) * C;
  if (i >= total) return;
  int row = (int)(i / C), c = (int)(i % C);
  if (row == n) {                       // zero pad-row for the next gather
    if (ybf) ybf[(size_t)row * bstride + boff + c] = (__bf16)0.f;
    return;
  }
  float mu  = stats[c] * invn;
  float var = stats[C + c] * invn - mu * mu;
  float v = (x[i] - mu) * rsqrtf(var + EPSV) * g[c] + b[c];
  if (resid) v += resid[i];
  if (relu)  v = fmaxf(v, 0.f);
  if (yf)  yf[i] = v;
  if (ybf) ybf[(size_t)row * bstride + boff + c] = (__bf16)v;
}

// ---------------- WMMA gather-GEMM-scatter conv ----------------
//
// fin : (Nin+1) x CIN bf16, row Nin all zeros (gather pad target)
// W   : gridDim.x x CIN x COUT f32 (row-major, per-offset)
// gidx/sidx : (nk, P) rulebook (pads = Nin / Nout respectively)
// out : (Nout+1) x COUT f32, pre-zeroed (row Nout = scatter dump row)
//
// One wave32 per 16-row tile. A-fragment per ISA 16-bit layout:
//   lane<16 : row M=lane,    K in {0..7, 16..23}   (two 16B chunks)
//   lane>=16: row M=lane-16, K in {8..15, 24..31}
// B-fragment mirrors with N = lane&15. C/D: vgpr r -> M = r + 8*(lane>>4).
template <int CIN, int COUT>
__global__ __launch_bounds__(256) void spconv_wmma_kernel(
    const __bf16* __restrict__ fin, const float* __restrict__ W,
    const int* __restrict__ gidx, const int* __restrict__ sidx,
    float* __restrict__ out, int P, int Nin, int Nout) {
  constexpr int KB = CIN / 32;   // K blocks of 32
  constexpr int NT = COUT / 16;  // N tiles of 16
  const int k      = blockIdx.x;
  const int lane   = threadIdx.x & 31;
  const int wv     = threadIdx.x >> 5;
  const int nwaves = blockDim.x >> 5;
  const int row16  = lane & 15;
  const int hf     = lane >> 4;

  const int*   gk = gidx + (size_t)k * P;
  const int*   sk = sidx + (size_t)k * P;
  const float* Wk = W + (size_t)k * CIN * COUT;

  // Hoist weight fragments into registers; reused for every tile this wave does.
  v16bf Bf[KB][NT];
#pragma unroll
  for (int kb = 0; kb < KB; kb++)
#pragma unroll
    for (int nt = 0; nt < NT; nt++)
#pragma unroll
      for (int e = 0; e < 16; e++) {
        int kk = ((e < 8) ? e : e + 8) + hf * 8 + kb * 32;     // K row in W
        Bf[kb][nt][e] = (__bf16)Wk[(size_t)kk * COUT + nt * 16 + row16];
      }

  const int ntiles = (P + 15) >> 4;
  for (int t = blockIdx.y * nwaves + wv; t < ntiles; t += (int)gridDim.y * nwaves) {
    // ---- gather A tile (16 rows x CIN bf16) ----
    int p  = t * 16 + row16;
    int gi = (p < P) ? gk[p] : Nin;               // pad -> zero row
    const __bf16* rowp = fin + (size_t)gi * CIN;

    v8f acc[NT];
#pragma unroll
    for (int nt = 0; nt < NT; nt++)
#pragma unroll
      for (int r = 0; r < 8; r++) acc[nt][r] = 0.f;

#pragma unroll
    for (int kb = 0; kb < KB; kb++) {
      v8bf lo = *(const v8bf*)(rowp + kb * 32 + hf * 8);        // K base..base+7
      v8bf hi = *(const v8bf*)(rowp + kb * 32 + 16 + hf * 8);   // K base+16..base+23
      v16bf a;
#pragma unroll
      for (int e = 0; e < 8; e++) { a[e] = lo[e]; a[e + 8] = hi[e]; }
#pragma unroll
      for (int nt = 0; nt < NT; nt++)
        acc[nt] = __builtin_amdgcn_wmma_f32_16x16x32_bf16(
            false, a, false, Bf[kb][nt], (short)0, acc[nt], false, false);
    }

    // ---- scatter-add D tile ----
    int orow[8];
#pragma unroll
    for (int r = 0; r < 8; r++) {
      int pp = t * 16 + hf * 8 + r;               // M = r + 8*hf
      orow[r] = (pp < P) ? sk[pp] : Nout;         // pad -> dump row
    }
#pragma unroll
    for (int nt = 0; nt < NT; nt++)
#pragma unroll
      for (int r = 0; r < 8; r++)
        atomicAdd(out + (size_t)orow[r] * COUT + nt * 16 + row16, acc[nt][r]);
  }
}

// ---------------- host orchestration ----------------

extern "C" void kernel_launch(void* const* d_in, const int* in_sizes, int n_in,
                              void* d_out, int out_size, void* d_ws, size_t ws_size,
                              hipStream_t stream) {
  (void)n_in; (void)out_size; (void)ws_size;

  const float* features = (const float*)d_in[0];
  const float* W_h1 = (const float*)d_in[1];  const float* g_h1 = (const float*)d_in[2];  const float* b_h1 = (const float*)d_in[3];
  const float* W_h2 = (const float*)d_in[4];  const float* g_h2 = (const float*)d_in[5];  const float* b_h2 = (const float*)d_in[6];
  const float* W_e  = (const float*)d_in[7];  const float* g_e  = (const float*)d_in[8];  const float* b_e  = (const float*)d_in[9];
  const float* W_u1 = (const float*)d_in[10]; const float* g_u1 = (const float*)d_in[11]; const float* b_u1 = (const float*)d_in[12];
  const float* W_u2 = (const float*)d_in[13]; const float* g_u2 = (const float*)d_in[14]; const float* b_u2 = (const float*)d_in[15];
  const float* W_d  = (const float*)d_in[16]; const float* g_d  = (const float*)d_in[17]; const float* b_d  = (const float*)d_in[18];
  const float* W_id = (const float*)d_in[19];
  const float* W_t1 = (const float*)d_in[20]; const float* g_t1 = (const float*)d_in[21]; const float* b_t1 = (const float*)d_in[22];
  const float* W_t2 = (const float*)d_in[23]; const float* g_t2 = (const float*)d_in[24]; const float* b_t2 = (const float*)d_in[25];
  const int* pin0  = (const int*)d_in[26];
  const int* pout0 = (const int*)d_in[27];
  const int* epin  = (const int*)d_in[28];
  const int* epout = (const int*)d_in[29];
  const int* pin1  = (const int*)d_in[30];
  const int* pout1 = (const int*)d_in[31];

  const int N0 = in_sizes[0] / 32;      // fine voxels
  const int N1 = in_sizes[28] / 27;     // coarse voxels (epin is 27 x N1)
  const size_t R0 = (size_t)N0 + 1, R1 = (size_t)N1 + 1;

  // ---- workspace carve-out (256B aligned slots) ----
  size_t off = 0;
  auto alloc = [&](size_t bytes) -> char* {
    char* p = (char*)d_ws + off;
    off += bytes; off = (off + 255) & ~(size_t)255;
    return p;
  };
  __bf16* bfA  = (__bf16*)alloc(R0 * 64 * sizeof(__bf16));  // x_bf / h1_bf / t1_bf
  __bf16* bfB  = (__bf16*)alloc(R0 * 64 * sizeof(__bf16));  // h_bf  / cat_bf (stride 64)
  __bf16* bfC  = (__bf16*)alloc(R1 * 64 * sizeof(__bf16));  // e_bf  / u_bf
  __bf16* bfD  = (__bf16*)alloc(R1 * 64 * sizeof(__bf16));  // u1_bf
  float*  fX   = (float*) alloc(R0 * 32 * sizeof(float));   // fine conv out (+dump row)
  float*  fY   = (float*) alloc((size_t)N0 * 32 * sizeof(float)); // h (f32, kept for concat)
  float*  fZ   = (float*) alloc(R0 * 32 * sizeof(float));   // idf (+dump row)
  float*  fE   = (float*) alloc(R1 * 64 * sizeof(float));   // coarse conv out (+dump row)
  float*  fF   = (float*) alloc((size_t)N1 * 64 * sizeof(float)); // e (f32, residual)
  float*  stats= (float*) alloc(128 * sizeof(float));
  int*    iota = (int*)   alloc((size_t)N0 * sizeof(int));

  const int TB = 256, GY = 64;
  auto g1 = [&](size_t n) { return dim3((unsigned)((n + TB - 1) / TB)); };
  auto zero = [&](float* p, size_t n) {
    zero_f32_kernel<<<dim3(1024), dim3(TB), 0, stream>>>(p, n);
  };
  const float i0 = 1.f / (float)N0, i1 = 1.f / (float)N1;

  // ---- Head resblock (fine, 32->32->32) ----
  cvt_bf_kernel<<<g1(R0 * 32), TB, 0, stream>>>(features, bfA, N0, 32, 32, 0);
  zero(fX, R0 * 32);
  spconv_wmma_kernel<32, 32><<<dim3(27, GY), TB, 0, stream>>>(bfA, W_h1, pin0, pout0, fX, N0, N0, N0);
  bn_stats_kernel<<<32, TB, 0, stream>>>(fX, N0, 32, stats);
  bn_apply_kernel<<<g1(R0 * 32), TB, 0, stream>>>(fX, stats, g_h1, b_h1, nullptr, nullptr, bfA, N0, 32, 32, 0, 1, i0);
  zero(fX, R0 * 32);
  spconv_wmma_kernel<32, 32><<<dim3(27, GY), TB, 0, stream>>>(bfA, W_h2, pin0, pout0, fX, N0, N0, N0);
  bn_stats_kernel<<<32, TB, 0, stream>>>(fX, N0, 32, stats);
  bn_apply_kernel<<<g1(R0 * 32), TB, 0, stream>>>(fX, stats, g_h2, b_h2, features, fY, bfB, N0, 32, 32, 0, 1, i0);

  // ---- Encode (stride-2, 32->64) ----
  zero(fE, R1 * 64);
  spconv_wmma_kernel<32, 64><<<dim3(27, GY), TB, 0, stream>>>(bfB, W_e, epin, epout, fE, N1, N0, N1);
  bn_stats_kernel<<<64, TB, 0, stream>>>(fE, N1, 64, stats);
  bn_apply_kernel<<<g1(R1 * 64), TB, 0, stream>>>(fE, stats, g_e, b_e, nullptr, fF, bfC, N1, 64, 64, 0, 1, i1);

  // ---- Inner resblock (coarse, 64->64->64) ----
  zero(fE, R1 * 64);
  spconv_wmma_kernel<64, 64><<<dim3(27, GY), TB, 0, stream>>>(bfC, W_u1, pin1, pout1, fE, N1, N1, N1);
  bn_stats_kernel<<<64, TB, 0, stream>>>(fE, N1, 64, stats);
  bn_apply_kernel<<<g1(R1 * 64), TB, 0, stream>>>(fE, stats, g_u1, b_u1, nullptr, nullptr, bfD, N1, 64, 64, 0, 1, i1);
  zero(fE, R1 * 64);
  spconv_wmma_kernel<64, 64><<<dim3(27, GY), TB, 0, stream>>>(bfD, W_u2, pin1, pout1, fE, N1, N1, N1);
  bn_stats_kernel<<<64, TB, 0, stream>>>(fE, N1, 64, stats);
  bn_apply_kernel<<<g1(R1 * 64), TB, 0, stream>>>(fE, stats, g_u2, b_u2, fF, nullptr, bfC, N1, 64, 64, 0, 1, i1);

  // ---- Decode (transposed rulebook, 64->32) ----
  zero(fX, R0 * 32);
  spconv_wmma_kernel<64, 32><<<dim3(27, GY), TB, 0, stream>>>(bfC, W_d, epout, epin, fX, N1, N1, N0);
  bn_stats_kernel<<<32, TB, 0, stream>>>(fX, N0, 32, stats);
  // build cat = [h | relu(bn(decode))] directly as bf16 (stride 64)
  cvt_bf_kernel<<<g1(R0 * 32), TB, 0, stream>>>(fY, bfB, N0, 32, 64, 0);
  bn_apply_kernel<<<g1(R0 * 32), TB, 0, stream>>>(fX, stats, g_d, b_d, nullptr, nullptr, bfB, N0, 32, 64, 32, 1, i0);

  // ---- identity path: idf = cat @ W_id (dense GEMM via iota rulebook, K=1) ----
  iota_kernel<<<g1((size_t)N0), TB, 0, stream>>>(iota, N0);
  zero(fZ, R0 * 32);
  spconv_wmma_kernel<64, 32><<<dim3(1, GY), TB, 0, stream>>>(bfB, W_id, iota, iota, fZ, N0, N0, N0);

  // ---- Tail resblock (64->32->32, residual = idf) ----
  zero(fX, R0 * 32);
  spconv_wmma_kernel<64, 32><<<dim3(27, GY), TB, 0, stream>>>(bfB, W_t1, pin0, pout0, fX, N0, N0, N0);
  bn_stats_kernel<<<32, TB, 0, stream>>>(fX, N0, 32, stats);
  bn_apply_kernel<<<g1(R0 * 32), TB, 0, stream>>>(fX, stats, g_t1, b_t1, nullptr, nullptr, bfA, N0, 32, 32, 0, 1, i0);
  zero(fX, R0 * 32);
  spconv_wmma_kernel<32, 32><<<dim3(27, GY), TB, 0, stream>>>(bfA, W_t2, pin0, pout0, fX, N0, N0, N0);
  bn_stats_kernel<<<32, TB, 0, stream>>>(fX, N0, 32, stats);
  bn_apply_kernel<<<g1(R0 * 32), TB, 0, stream>>>(fX, stats, g_t2, b_t2, fZ, (float*)d_out, nullptr, N0, 32, 32, 0, 1, i0);
}
